// MoEContainer_75977971466567
// MI455X (gfx1250) — matrix operand order repaired and checked by reading, not compiled, not run on hardware
//
#include <hip/hip_runtime.h>
#include <hip/hip_bf16.h>

// ---- problem sizes (B=2, T=2048, D=1024, E=8, H=4096, top-K=2) ----
#define NTOK  4096
#define DIM   1024
#define NEXP  8
#define HID   4096
#define KSEL  2
#define NPAIR (NTOK*KSEL)

// ---- GEMM tiling: 128x128x32, 256 threads (8 wave32), wave tile 64x32 ----
#define BM 128
#define BN 128
#define BK 32
#define THREADS 256
#define SMEM_BYTES ((BM*BK + BK*BN) * 2)   // 16 KB of the 320 KB WGP pool

typedef __attribute__((ext_vector_type(16))) __bf16 v16bf;
typedef __attribute__((ext_vector_type(8)))  float  v8f;
typedef unsigned int u32x4 __attribute__((ext_vector_type(4)));
typedef int          i32x4 __attribute__((ext_vector_type(4)));
typedef int          i32x8 __attribute__((ext_vector_type(8)));

// native bf16 converts -> v_cvt_pk_bf16_f32 on gfx1250 (RTNE in hw)
static __device__ __forceinline__ __bf16 f2bf(float f) { return (__bf16)f; }
static __device__ __forceinline__ float  bf2f(__bf16 b) { return (float)b; }
static __device__ __forceinline__ float silu_f(float v) {
  return v / (1.0f + __expf(-v));
}

// ------------- TDM: async 2D tile load (global bf16 -> LDS) -----------------
// D# per ISA cdna5 §8.3/8.4: data_size=8B units; tile = tile_dim0 x tile_dim1;
// rows beyond tensor_dim1 are hardware zero-filled in LDS (ragged expert tail).
static __device__ __forceinline__ void tdm_load_a_tile(unsigned lds_off,
                                                       const __bf16* gaddr,
                                                       unsigned row_units,
                                                       unsigned rows_avail) {
  const unsigned long long ga = (unsigned long long)(size_t)gaddr;
  u32x4 g0;
  g0[0] = 1u;                                            // count=1 valid D#
  g0[1] = lds_off;                                       // LDS byte address
  g0[2] = (unsigned)(ga & 0xFFFFFFFFu);                  // global addr [31:0]
  g0[3] = (unsigned)((ga >> 32) & 0x01FFFFFFu) | 0x80000000u; // [56:32]|type=2
  i32x8 g1;
  g1[0] = 0x00030000;                   // workgroup_mask=0, data_size=3 (8B)
  g1[1] = (int)(8u << 16);              // tensor_dim0 = 8 units (64B row)
  g1[2] = (int)((rows_avail & 0xFFFFu) << 16);           // tensor_dim1 lo16
  g1[3] = (int)(((rows_avail >> 16) & 0xFFFFu) | (8u << 16)); // dim1 hi|tile_dim0=8
  g1[4] = (int)BM;                      // tile_dim1 = 128 rows, tile_dim2 = 0
  g1[5] = (int)row_units;               // tensor_dim0_stride (lo32), units of 8B
  g1[6] = 0;                            // stride hi16 | tensor_dim1_stride lo
  g1[7] = 0;
  const i32x4 z4 = {0, 0, 0, 0};
#if __clang_major__ >= 23
  const i32x8 z8 = {0, 0, 0, 0, 0, 0, 0, 0};
  __builtin_amdgcn_tensor_load_to_lds(g0, g1, z4, z4, z8, 0);
#else
  __builtin_amdgcn_tensor_load_to_lds(g0, g1, z4, z4, 0);
#endif
}

// ---------------- WMMA core: one (As,Bs) K-step for one wave ----------------
// A frag (16x32): lane&15 = M row; lanes<16 hold K {0..7,16..23}, lanes>=16
// hold K {8..15,24..31} (ISA 7.12.2). B frag (32x16): lane&15 = N col, same
// K order as A. C/D: element r at lane L -> (M = r + 8*(L>>4), N = L&15).
static __device__ __forceinline__ void mma_tiles(const __bf16* As, const __bf16* Bs,
                                                 v8f acc[4][2], int lane,
                                                 int m0, int n0) {
  const int lhalf = lane >> 4, lmod = lane & 15;
  const int ka = lhalf * 8;
  v16bf afrag[4];
#pragma unroll
  for (int i = 0; i < 4; ++i) {
    const __bf16* p = As + (size_t)(m0 + i * 16 + lmod) * BK;
    union { uint4 q[2]; v16bf v; } u;
    u.q[0] = *(const uint4*)(p + ka);
    u.q[1] = *(const uint4*)(p + ka + 16);
    afrag[i] = u.v;
  }
  const int kb = lhalf * 8;
  v16bf bfrag[2];
#pragma unroll
  for (int j = 0; j < 2; ++j) {
    const __bf16* p = Bs + n0 + j * 16 + lmod;
    v16bf v;
#pragma unroll
    for (int s = 0; s < 8; ++s) {
      v[s] = p[(size_t)(kb + s) * BN];
      v[s + 8] = p[(size_t)(kb + 16 + s) * BN];
    }
    bfrag[j] = v;
  }
#pragma unroll
  for (int i = 0; i < 4; ++i)
#pragma unroll
    for (int j = 0; j < 2; ++j)
      acc[i][j] = __builtin_amdgcn_wmma_f32_16x16x32_bf16(
          false, afrag[i], false, bfrag[j], (short)0, acc[i][j], false, false);
}

// B: fp32 weights [K][N] row-major -> bf16 LDS tile [BK][BN] (256 threads).
static __device__ __forceinline__ void load_b_tile(__bf16* Bs, const float* W,
                                                   int ldw, int krow0, int ncol0,
                                                   int tid) {
  const int kr = tid >> 3;            // 0..31
  const int ns = (tid & 7) * 16;      // 0,16,...,112
  const float4* src = (const float4*)(W + (size_t)(krow0 + kr) * ldw + ncol0 + ns);
  union { float4 f4[4]; float f[16]; } fin;
#pragma unroll
  for (int i = 0; i < 4; ++i) fin.f4[i] = src[i];
  union { __bf16 e[16]; uint4 q[2]; } u;
#pragma unroll
  for (int i = 0; i < 16; ++i) u.e[i] = f2bf(fin.f[i]);
  uint4* dst = (uint4*)(Bs + (size_t)kr * BN + ns);
  dst[0] = u.q[0]; dst[1] = u.q[1];
}

// A: bf16 rows gathered through a token-index list; zero-fill pad rows.
static __device__ __forceinline__ void load_a_gather(__bf16* As, const __bf16* X,
                                                     int lda, const int* toks,
                                                     int nrows, int mrow0,
                                                     int kcol0, int tid) {
  const int r = tid >> 1;
  const int h = (tid & 1) * 16;
  uint4 v0 = make_uint4(0, 0, 0, 0), v1 = make_uint4(0, 0, 0, 0);
  const int p = mrow0 + r;
  if (p < nrows) {
    const int tok = toks[p];
    const uint4* src = (const uint4*)(X + (size_t)tok * lda + kcol0 + h);
    v0 = src[0]; v1 = src[1];
  }
  uint4* dst = (uint4*)(As + (size_t)r * BK + h);
  dst[0] = v0; dst[1] = v1;
}

// ---------------- kernel 0: xb = bf16(x), zero expert counts ----------------
__global__ __launch_bounds__(THREADS) void k_init(const float* __restrict__ x,
                                                  __bf16* __restrict__ xb,
                                                  int* __restrict__ counts) {
  const int gid = blockIdx.x * THREADS + threadIdx.x;   // over NTOK*DIM/4
  const float4 v = ((const float4*)x)[gid];
  union { __bf16 e[4]; uint2 q; } u;
  u.e[0] = f2bf(v.x); u.e[1] = f2bf(v.y); u.e[2] = f2bf(v.z); u.e[3] = f2bf(v.w);
  ((uint2*)xb)[gid] = u.q;
  if (gid < NEXP) counts[gid] = 0;
}

// ---------------- kernel 1: router GEMM1  t = silu(x @ rw1 + rb1) -----------
// A tile via TDM (tensor_load_to_lds), B tile via VGPR path with bf16 cvt.
__global__ __launch_bounds__(THREADS) void k_router1(const __bf16* __restrict__ xb,
                                                     const float* __restrict__ rw1,
                                                     const float* __restrict__ rb1,
                                                     __bf16* __restrict__ t) {
  extern __shared__ __align__(16) __bf16 smem[];
  __bf16* As = smem;                 // LDS offset 0, BM*BK bf16
  __bf16* Bs = smem + BM * BK;       // LDS offset 8192, BK*BN bf16
  const unsigned as_off = (unsigned)(size_t)(void*)As;
  const int tid = threadIdx.x, lane = tid & 31, wave = tid >> 5;
  const int m0 = (wave >> 2) * 64, n0 = (wave & 3) * 32;
  const int mrow0 = blockIdx.x * BM, ncol0 = blockIdx.y * BN;
  v8f acc[4][2] = {};
  for (int kc = 0; kc < DIM; kc += BK) {
    __syncthreads();                                  // prev MMA done
    if (wave == 0)
      tdm_load_a_tile(as_off, xb + (size_t)mrow0 * DIM + kc,
                      DIM * 2 / 8, (unsigned)(NTOK - mrow0));
    load_b_tile(Bs, rw1, DIM, kc, ncol0, tid);
    if (kc + BK < DIM)                                // gfx1250 global_prefetch
      __builtin_prefetch(rw1 + (size_t)(kc + BK + (tid >> 3)) * DIM + ncol0 + (tid & 7) * 16);
    if (wave == 0) __builtin_amdgcn_s_wait_tensorcnt(0);
    __syncthreads();                                  // tiles visible
    mma_tiles(As, Bs, acc, lane, m0, n0);
  }
  const int lhalf = lane >> 4, lmod = lane & 15;
#pragma unroll
  for (int j = 0; j < 2; ++j) {
    const int col = ncol0 + n0 + j * 16 + lmod;
    const float bias = rb1[col];
#pragma unroll
    for (int i = 0; i < 4; ++i) {
#pragma unroll
      for (int r = 0; r < 8; ++r) {
        const int row = mrow0 + m0 + i * 16 + r + lhalf * 8;
        t[(size_t)row * DIM + col] = f2bf(silu_f(acc[i][j][r] + bias));
      }
    }
  }
}

// ---------------- kernel 2: router GEMM2  logits = t @ rw2 + rb2 (E=8) ------
__global__ __launch_bounds__(THREADS) void k_router2(const __bf16* __restrict__ t,
                                                     const float* __restrict__ rw2,
                                                     const float* __restrict__ rb2,
                                                     float* __restrict__ logits) {
  const int tid = blockIdx.x * THREADS + threadIdx.x;   // over NTOK*NEXP
  const int n = tid >> 3, e = tid & 7;
  float s = 0.0f;
  const __bf16* row = t + (size_t)n * DIM;
  for (int k = 0; k < DIM; ++k) s += bf2f(row[k]) * rw2[(size_t)k * NEXP + e];
  logits[tid] = s + rb2[e];
}

// ---------------- kernel 3: top-2 + masked softmax weights ------------------
__global__ __launch_bounds__(THREADS) void k_topk(const float* __restrict__ logits,
                                                  int* __restrict__ tokE,
                                                  float* __restrict__ tokW,
                                                  int* __restrict__ counts) {
  const int n = blockIdx.x * THREADS + threadIdx.x;
  if (n >= NTOK) return;
  float l[NEXP];
#pragma unroll
  for (int e = 0; e < NEXP; ++e) l[e] = logits[n * NEXP + e];
  int i0 = 0;
#pragma unroll
  for (int e = 1; e < NEXP; ++e) if (l[e] > l[i0]) i0 = e;
  int i1 = -1;
#pragma unroll
  for (int e = 0; e < NEXP; ++e)
    if (e != i0 && (i1 < 0 || l[e] > l[i1])) i1 = e;
  const float e1 = __expf(l[i1] - l[i0]);     // softmax over the two selected
  const float inv = 1.0f / (1.0f + e1);
  tokE[n * 2 + 0] = i0; tokE[n * 2 + 1] = i1;
  tokW[n * 2 + 0] = inv; tokW[n * 2 + 1] = e1 * inv;
  atomicAdd(&counts[i0], 1);
  atomicAdd(&counts[i1], 1);
}

// ---------------- kernel 4: exclusive prefix over E=8 counts ----------------
__global__ void k_offsets(const int* __restrict__ counts, int* __restrict__ offsets,
                          int* __restrict__ cnt2) {
  if (threadIdx.x == 0) {
    int a = 0;
#pragma unroll
    for (int e = 0; e < NEXP; ++e) { offsets[e] = a; a += counts[e]; cnt2[e] = 0; }
  }
}

// ---------------- kernel 5: build compact per-expert token lists ------------
__global__ __launch_bounds__(THREADS) void k_scatter(const int* __restrict__ tokE,
                                                     const int* __restrict__ offsets,
                                                     int* __restrict__ cnt2,
                                                     int* __restrict__ pairTok,
                                                     int* __restrict__ tokPos) {
  const int n = blockIdx.x * THREADS + threadIdx.x;
  if (n >= NTOK) return;
#pragma unroll
  for (int j = 0; j < KSEL; ++j) {
    const int e = tokE[n * 2 + j];
    const int pos = offsets[e] + atomicAdd(&cnt2[e], 1);
    pairTok[pos] = n;
    tokPos[n * 2 + j] = pos;
  }
}

// ---------------- kernel 6: expert FFN1  h = silu(gather(x) @ we1 + be1) ----
// A rows are gathered per-token -> VGPR gather path (TDM gather mode caps at
// 16 indexed rows, unusable for 128-row tiles).
__global__ __launch_bounds__(THREADS) void k_ffn1(const __bf16* __restrict__ xb,
                                                  const float* __restrict__ we1,
                                                  const float* __restrict__ be1,
                                                  const int* __restrict__ counts,
                                                  const int* __restrict__ offsets,
                                                  const int* __restrict__ pairTok,
                                                  __bf16* __restrict__ hbuf) {
  const int e = blockIdx.z;
  const int count = counts[e];
  const int mrow0 = blockIdx.x * BM;
  if (mrow0 >= count) return;
  const int base = offsets[e];
  const float* W = we1 + (size_t)e * DIM * HID;
  const float* bias = be1 + (size_t)e * HID;
  const int* toks = pairTok + base;

  extern __shared__ __align__(16) __bf16 smem[];
  __bf16* As = smem;
  __bf16* Bs = smem + BM * BK;
  const int tid = threadIdx.x, lane = tid & 31, wave = tid >> 5;
  const int m0 = (wave >> 2) * 64, n0 = (wave & 3) * 32;
  const int ncol0 = blockIdx.y * BN;
  v8f acc[4][2] = {};
  for (int kc = 0; kc < DIM; kc += BK) {
    __syncthreads();
    load_a_gather(As, xb, DIM, toks, count, mrow0, kc, tid);
    load_b_tile(Bs, W, HID, kc, ncol0, tid);
    if (kc + BK < DIM)
      __builtin_prefetch(W + (size_t)(kc + BK + (tid >> 3)) * HID + ncol0 + (tid & 7) * 16);
    __syncthreads();
    mma_tiles(As, Bs, acc, lane, m0, n0);
  }
  const int lhalf = lane >> 4, lmod = lane & 15;
#pragma unroll
  for (int j = 0; j < 2; ++j) {
    const int col = ncol0 + n0 + j * 16 + lmod;
    const float b = bias[col];
#pragma unroll
    for (int i = 0; i < 4; ++i) {
#pragma unroll
      for (int r = 0; r < 8; ++r) {
        const int p = mrow0 + m0 + i * 16 + r + lhalf * 8;   // expert-local row
        if (p < count)
          hbuf[(size_t)(base + p) * HID + col] = f2bf(silu_f(acc[i][j][r] + b));
      }
    }
  }
}

// ---------------- kernel 7: expert FFN2  eo = h @ we2 + be2 (compact rows) --
// A tile via TDM; tensor_dim1 = count-mrow0 so the ragged tail rows are
// hardware zero-filled in LDS.
__global__ __launch_bounds__(THREADS) void k_ffn2(const __bf16* __restrict__ hbuf,
                                                  const float* __restrict__ we2,
                                                  const float* __restrict__ be2,
                                                  const int* __restrict__ counts,
                                                  const int* __restrict__ offsets,
                                                  float* __restrict__ eo) {
  const int e = blockIdx.z;
  const int count = counts[e];
  const int mrow0 = blockIdx.x * BM;
  if (mrow0 >= count) return;
  const int base = offsets[e];
  const float* W = we2 + (size_t)e * HID * DIM;
  const float* bias = be2 + (size_t)e * DIM;
  const __bf16* A = hbuf + (size_t)base * HID;

  extern __shared__ __align__(16) __bf16 smem[];
  __bf16* As = smem;
  __bf16* Bs = smem + BM * BK;
  const unsigned as_off = (unsigned)(size_t)(void*)As;
  const int tid = threadIdx.x, lane = tid & 31, wave = tid >> 5;
  const int m0 = (wave >> 2) * 64, n0 = (wave & 3) * 32;
  const int ncol0 = blockIdx.y * BN;
  v8f acc[4][2] = {};
  for (int kc = 0; kc < HID; kc += BK) {
    __syncthreads();
    if (wave == 0)
      tdm_load_a_tile(as_off, A + (size_t)mrow0 * HID + kc,
                      HID * 2 / 8, (unsigned)(count - mrow0));
    load_b_tile(Bs, W, DIM, kc, ncol0, tid);
    if (kc + BK < HID)
      __builtin_prefetch(W + (size_t)(kc + BK + (tid >> 3)) * DIM + ncol0 + (tid & 7) * 16);
    if (wave == 0) __builtin_amdgcn_s_wait_tensorcnt(0);
    __syncthreads();
    mma_tiles(As, Bs, acc, lane, m0, n0);
  }
  const int lhalf = lane >> 4, lmod = lane & 15;
#pragma unroll
  for (int j = 0; j < 2; ++j) {
    const int col = ncol0 + n0 + j * 16 + lmod;
    const float b = bias[col];
#pragma unroll
    for (int i = 0; i < 4; ++i) {
#pragma unroll
      for (int r = 0; r < 8; ++r) {
        const int p = mrow0 + m0 + i * 16 + r + lhalf * 8;
        if (p < count)
          eo[(size_t)(base + p) * DIM + col] = acc[i][j][r] + b;
      }
    }
  }
}

// ---------------- kernel 8: deterministic combine  y = x + w0*eo0 + w1*eo1 --
__global__ __launch_bounds__(THREADS) void k_combine(const float* __restrict__ x,
                                                     const float* __restrict__ eo,
                                                     const int* __restrict__ tokPos,
                                                     const float* __restrict__ tokW,
                                                     float* __restrict__ y) {
  const int gid = blockIdx.x * THREADS + threadIdx.x;   // over NTOK*(DIM/4)
  const int n = gid >> 8;                               // DIM/4 == 256
  const int c = gid & 255;
  const int p0 = tokPos[n * 2 + 0], p1 = tokPos[n * 2 + 1];
  const float w0 = tokW[n * 2 + 0], w1 = tokW[n * 2 + 1];
  const float4 a  = ((const float4*)x)[gid];
  const float4 b0 = ((const float4*)eo)[(size_t)p0 * 256 + c];
  const float4 b1 = ((const float4*)eo)[(size_t)p1 * 256 + c];
  float4 r;
  r.x = a.x + w0 * b0.x + w1 * b1.x;
  r.y = a.y + w0 * b0.y + w1 * b1.y;
  r.z = a.z + w0 * b0.z + w1 * b1.z;
  r.w = a.w + w0 * b0.w + w1 * b1.w;
  ((float4*)y)[gid] = r;
}

// ------------------------------- launcher -----------------------------------
extern "C" void kernel_launch(void* const* d_in, const int* in_sizes, int n_in,
                              void* d_out, int out_size, void* d_ws, size_t ws_size,
                              hipStream_t stream) {
  (void)in_sizes; (void)n_in; (void)out_size; (void)ws_size;
  const float* x   = (const float*)d_in[0];
  const float* rw1 = (const float*)d_in[1];
  const float* rb1 = (const float*)d_in[2];
  const float* rw2 = (const float*)d_in[3];
  const float* rb2 = (const float*)d_in[4];
  const float* we1 = (const float*)d_in[5];
  const float* be1 = (const float*)d_in[6];
  const float* we2 = (const float*)d_in[7];
  const float* be2 = (const float*)d_in[8];
  float* y = (float*)d_out;

  // workspace carve-out (all offsets 64B-aligned)
  char* ws = (char*)d_ws;
  size_t off = 0;
  __bf16* xb      = (__bf16*)(ws + off); off += (size_t)NTOK * DIM * 2;   // 8 MB
  __bf16* t       = (__bf16*)(ws + off); off += (size_t)NTOK * DIM * 2;   // 8 MB
  float*  logits  = (float*) (ws + off); off += (size_t)NTOK * NEXP * 4;
  int*    tokE    = (int*)   (ws + off); off += (size_t)NTOK * KSEL * 4;
  float*  tokW    = (float*) (ws + off); off += (size_t)NTOK * KSEL * 4;
  int*    tokPos  = (int*)   (ws + off); off += (size_t)NTOK * KSEL * 4;
  int*    counts  = (int*)   (ws + off); off += 64;
  int*    offsets = (int*)   (ws + off); off += 64;
  int*    cnt2    = (int*)   (ws + off); off += 64;
  int*    pairTok = (int*)   (ws + off); off += (size_t)NPAIR * 4;
  __bf16* hbuf    = (__bf16*)(ws + off); off += (size_t)NPAIR * HID * 2;  // 64 MB
  float*  eo      = (float*) (ws + off); off += (size_t)NPAIR * DIM * 4;  // 32 MB

  k_init   <<<dim3(NTOK * DIM / 4 / THREADS), dim3(THREADS), 0, stream>>>(x, xb, counts);
  k_router1<<<dim3(NTOK / BM, DIM / BN),      dim3(THREADS), SMEM_BYTES, stream>>>(xb, rw1, rb1, t);
  k_router2<<<dim3(NTOK * NEXP / THREADS),    dim3(THREADS), 0, stream>>>(t, rw2, rb2, logits);
  k_topk   <<<dim3(NTOK / THREADS),           dim3(THREADS), 0, stream>>>(logits, tokE, tokW, counts);
  k_offsets<<<dim3(1), dim3(32), 0, stream>>>(counts, offsets, cnt2);
  k_scatter<<<dim3(NTOK / THREADS),           dim3(THREADS), 0, stream>>>(tokE, offsets, cnt2, pairTok, tokPos);
  k_ffn1   <<<dim3(NTOK / BM, HID / BN, NEXP), dim3(THREADS), SMEM_BYTES, stream>>>(xb, we1, be1, counts, offsets, pairTok, hbuf);
  k_ffn2   <<<dim3(NTOK / BM, DIM / BN, NEXP), dim3(THREADS), SMEM_BYTES, stream>>>(hbuf, we2, be2, counts, offsets, eo);
  k_combine<<<dim3(NTOK * DIM / 4 / THREADS), dim3(THREADS), 0, stream>>>(x, eo, tokPos, tokW, y);
}
